// model_reconstruct_47974784697104
// MI455X (gfx1250) — matrix operand order repaired and checked by reading, not compiled, not run on hardware
//
#include <hip/hip_runtime.h>
#include <hip/hip_bf16.h>

typedef __bf16 bf16_t;
typedef __attribute__((ext_vector_type(16))) __bf16        v16bf;
typedef __attribute__((ext_vector_type(4)))  __bf16        v4bf;
typedef __attribute__((ext_vector_type(8)))  float         v8f;
typedef __attribute__((ext_vector_type(4)))  unsigned int  u32x4;

// POD fragment: two 16B chunks alias the 32B WMMA operand -> no element shuffles
union frag16 { u32x4 q[2]; v16bf v; };

#define N_ROWS 4096
#define M_ALL  8192          // v1 and v2 stacked
#define DIM    512
#define TAU_INV 1.25f        // 1 / 0.8

// ---------------------------------------------------------------------------
// CDNA5 async global->LDS copy (ASYNCcnt path, ISA 10.7 / 15.18.3).
// LDS byte offset = low 32 bits of the generic pointer (LDS aperture tag is in
// ADDR[63:32] per ISA 10.2 aperture check, low bits are the LDS offset).
// ---------------------------------------------------------------------------
__device__ __forceinline__ void async_g2l_b128(void* lds, const void* gptr) {
    const unsigned loff = (unsigned)(unsigned long long)lds;
    asm volatile("global_load_async_to_lds_b128 %0, %1, off"
                 :: "v"(loff), "v"(gptr) : "memory");
}
__device__ __forceinline__ void wait_async0() {
    asm volatile("s_wait_asynccnt 0x0" ::: "memory");
}

// ---------------------------------------------------------------------------
// Kernel 0: elementwise f32 -> bf16 (4 elems / thread)
// ---------------------------------------------------------------------------
__global__ __launch_bounds__(256) void cvt_bf16_kernel(
    const float* __restrict__ src, bf16_t* __restrict__ dst, int n4)
{
    const int i = blockIdx.x * 256 + threadIdx.x;
    if (i < n4) {
        const float4 f = ((const float4*)src)[i];
        v4bf o;
        o[0] = (bf16_t)f.x; o[1] = (bf16_t)f.y;
        o[2] = (bf16_t)f.z; o[3] = (bf16_t)f.w;
        ((v4bf*)dst)[i] = o;
    }
}

// ---------------------------------------------------------------------------
// Kernel 1: Z = ELU(Vb @ Wb^T + bias) for a 128x128 tile; also emits
// per-(row, col-block) sum-of-squares partials (deterministic).
// Double-buffered LDS NT GEMM with async global->LDS staging.
// grid = (DIM/128 = 4, M_ALL/128 = 64)
// WMMA layouts (ISA 7.12.2):
//   A 16x32 bf16 : lane = kh*16+m ; elems 0-7 -> K=kh*8..+7, 8-15 -> K=kh*8+16..+7
//   B 32x16 bf16 : lane = kh*16+n ; elems 0-7 -> K=kh*8..+7, 8-15 -> K=kh*8+16..+7   (B[k][n] = W[n][k])
//   C 16x16 f32  : elem r of lane -> m = r+8*kh, n = lane&15
// ---------------------------------------------------------------------------
__global__ __launch_bounds__(256) void gemm_bias_elu_kernel(
    const bf16_t* __restrict__ Vb,   // [M_ALL][DIM]
    const bf16_t* __restrict__ Wb,   // [DIM][DIM]
    const float*  __restrict__ bias, // [DIM]
    bf16_t* __restrict__ Zb,         // [M_ALL][DIM]
    float*  __restrict__ Zssq)       // [M_ALL][4]
{
    __shared__ __align__(16) bf16_t As[2][128][40];  // 80B row stride, 16B aligned
    __shared__ __align__(16) bf16_t Bs[2][128][40];
    __shared__ float partial[128][17];

    const int wave = threadIdx.x >> 5;
    const int lane = threadIdx.x & 31;
    const int m16  = lane & 15;
    const int kh   = lane >> 4;
    const int rowBase = blockIdx.y * 128;
    const int colBase = blockIdx.x * 128;

    v8f acc[8];
#pragma unroll
    for (int t = 0; t < 8; ++t) {
#pragma unroll
        for (int e = 0; e < 8; ++e) acc[t][e] = 0.0f;
    }

    const int ldRow = threadIdx.x >> 1;
    const int ldCol = (threadIdx.x & 1) * 16;
    const bf16_t* g1 = Vb + (size_t)(rowBase + ldRow) * DIM + ldCol;
    const bf16_t* g2 = Wb + (size_t)(colBase + ldRow) * DIM + ldCol;

    // prologue: async-stage K-step 0 into buffer 0
    async_g2l_b128(&As[0][ldRow][ldCol],     g1);
    async_g2l_b128(&As[0][ldRow][ldCol + 8], g1 + 8);
    async_g2l_b128(&Bs[0][ldRow][ldCol],     g2);
    async_g2l_b128(&Bs[0][ldRow][ldCol + 8], g2 + 8);
    wait_async0();
    __syncthreads();

    int buf = 0;
#pragma unroll
    for (int k0 = 0; k0 < DIM; k0 += 32) {
        const bool has_next = (k0 + 32) < DIM;
        if (has_next) {  // issue next tile's async copies; drain during WMMAs
            const int nb = buf ^ 1;
            async_g2l_b128(&As[nb][ldRow][ldCol],     g1 + k0 + 32);
            async_g2l_b128(&As[nb][ldRow][ldCol + 8], g1 + k0 + 40);
            async_g2l_b128(&Bs[nb][ldRow][ldCol],     g2 + k0 + 32);
            async_g2l_b128(&Bs[nb][ldRow][ldCol + 8], g2 + k0 + 40);
        }

        frag16 fa;
        {
            const bf16_t* ap = &As[buf][wave * 16 + m16][kh * 8];
            fa.q[0] = *(const u32x4*)ap;
            fa.q[1] = *(const u32x4*)(ap + 16);
        }
#pragma unroll
        for (int t = 0; t < 8; ++t) {
            frag16 fb;
            const bf16_t* bp = &Bs[buf][t * 16 + m16][kh * 8];
            fb.q[0] = *(const u32x4*)bp;
            fb.q[1] = *(const u32x4*)(bp + 16);
            acc[t] = __builtin_amdgcn_wmma_f32_16x16x32_bf16(
                false, fa.v, false, fb.v, (short)0, acc[t], false, false);
        }

        if (has_next) {
            wait_async0();      // this wave's async writes to buf^1 landed
            __syncthreads();    // all waves' writes visible; readers of buf done
            buf ^= 1;
        }
    }

    // fused epilogue: bias + ELU, store bf16 z, accumulate row sum-of-squares
    float ssq[8];
#pragma unroll
    for (int r = 0; r < 8; ++r) ssq[r] = 0.0f;

#pragma unroll
    for (int t = 0; t < 8; ++t) {
        const int n  = colBase + t * 16 + m16;
        const float bv = bias[n];
#pragma unroll
        for (int r = 0; r < 8; ++r) {
            float x = acc[t][r] + bv;
            x = (x > 0.0f) ? x : expm1f(x);
            const int i = rowBase + wave * 16 + r + 8 * kh;
            Zb[(size_t)i * DIM + n] = (bf16_t)x;
            ssq[r] = fmaf(x, x, ssq[r]);
        }
    }
#pragma unroll
    for (int r = 0; r < 8; ++r)
        partial[wave * 16 + r + 8 * kh][m16] = ssq[r];
    __syncthreads();

    if (threadIdx.x < 128) {
        float s = 0.0f;
#pragma unroll
        for (int c = 0; c < 16; ++c) s += partial[threadIdx.x][c];
        Zssq[(size_t)(rowBase + threadIdx.x) * 4 + blockIdx.x] = s;
    }
}

// ---------------------------------------------------------------------------
// Kernel 2: per-row rnorm from the 4 col-block partials; scale row in place.
// grid = (M_ALL), block = 128 (4 bf16 per thread covers DIM=512)
// ---------------------------------------------------------------------------
__global__ __launch_bounds__(128) void norm_scale_kernel(
    bf16_t* __restrict__ Zb, const float* __restrict__ Zssq)
{
    const int row = blockIdx.x;
    const float s = Zssq[(size_t)row * 4 + 0] + Zssq[(size_t)row * 4 + 1]
                  + Zssq[(size_t)row * 4 + 2] + Zssq[(size_t)row * 4 + 3];
    const float rn = rsqrtf(s + 1e-12f);
    v4bf* p = (v4bf*)(Zb + (size_t)row * DIM) + threadIdx.x;
    v4bf v = *p;
#pragma unroll
    for (int e = 0; e < 4; ++e) v[e] = (bf16_t)((float)v[e] * rn);
    *p = v;
}

// ---------------------------------------------------------------------------
// Kernel 3: fused  sim = exp((zhat1 @ zhat2^T) / tau);  sp += sim*pos;
// st += sim*(pos+neg).  Async-staged double-buffered LDS NT GEMM +
// pos/neg L2 prefetch so the 128MB HBM term overlaps the GEMM.
// ---------------------------------------------------------------------------
__global__ __launch_bounds__(256) void sim_reduce_kernel(
    const bf16_t* __restrict__ Z1, const bf16_t* __restrict__ Z2,
    const float* __restrict__ pos, const float* __restrict__ neg,
    float* __restrict__ partials)
{
    __shared__ __align__(16) bf16_t As[2][128][40];
    __shared__ __align__(16) bf16_t Bs[2][128][40];
    __shared__ float red[16];

    const int wave = threadIdx.x >> 5;
    const int lane = threadIdx.x & 31;
    const int m16  = lane & 15;
    const int kh   = lane >> 4;
    const int rowBase = blockIdx.y * 128;
    const int colBase = blockIdx.x * 128;

    // Prefetch this block's 128KB pos/neg tile into L2 (overlap with GEMM).
    {
        const int prow = threadIdx.x >> 1;
        const size_t pbase = (size_t)(rowBase + prow) * N_ROWS + colBase
                           + (size_t)(threadIdx.x & 1) * 64;
        __builtin_prefetch(pos + pbase,      0, 0);
        __builtin_prefetch(pos + pbase + 32, 0, 0);
        __builtin_prefetch(neg + pbase,      0, 0);
        __builtin_prefetch(neg + pbase + 32, 0, 0);
    }

    v8f acc[8];
#pragma unroll
    for (int t = 0; t < 8; ++t) {
#pragma unroll
        for (int e = 0; e < 8; ++e) acc[t][e] = 0.0f;
    }

    const int ldRow = threadIdx.x >> 1;
    const int ldCol = (threadIdx.x & 1) * 16;
    const bf16_t* g1 = Z1 + (size_t)(rowBase + ldRow) * DIM + ldCol;
    const bf16_t* g2 = Z2 + (size_t)(colBase + ldRow) * DIM + ldCol;

    // prologue: async-stage K-step 0 into buffer 0
    async_g2l_b128(&As[0][ldRow][ldCol],     g1);
    async_g2l_b128(&As[0][ldRow][ldCol + 8], g1 + 8);
    async_g2l_b128(&Bs[0][ldRow][ldCol],     g2);
    async_g2l_b128(&Bs[0][ldRow][ldCol + 8], g2 + 8);
    wait_async0();
    __syncthreads();

    int buf = 0;
#pragma unroll
    for (int k0 = 0; k0 < DIM; k0 += 32) {
        const bool has_next = (k0 + 32) < DIM;
        if (has_next) {
            const int nb = buf ^ 1;
            async_g2l_b128(&As[nb][ldRow][ldCol],     g1 + k0 + 32);
            async_g2l_b128(&As[nb][ldRow][ldCol + 8], g1 + k0 + 40);
            async_g2l_b128(&Bs[nb][ldRow][ldCol],     g2 + k0 + 32);
            async_g2l_b128(&Bs[nb][ldRow][ldCol + 8], g2 + k0 + 40);
        }

        frag16 fa;
        {
            const bf16_t* ap = &As[buf][wave * 16 + m16][kh * 8];
            fa.q[0] = *(const u32x4*)ap;
            fa.q[1] = *(const u32x4*)(ap + 16);
        }
#pragma unroll
        for (int t = 0; t < 8; ++t) {
            frag16 fb;
            const bf16_t* bp = &Bs[buf][t * 16 + m16][kh * 8];
            fb.q[0] = *(const u32x4*)bp;
            fb.q[1] = *(const u32x4*)(bp + 16);
            acc[t] = __builtin_amdgcn_wmma_f32_16x16x32_bf16(
                false, fa.v, false, fb.v, (short)0, acc[t], false, false);
        }

        if (has_next) {
            wait_async0();
            __syncthreads();
            buf ^= 1;
        }
    }

    // fused epilogue: sim = exp(cos/tau), mask, local accumulate
    float sp = 0.0f, st = 0.0f;
#pragma unroll
    for (int t = 0; t < 8; ++t) {
        const int j = colBase + t * 16 + m16;
#pragma unroll
        for (int r = 0; r < 8; ++r) {
            const int i = rowBase + wave * 16 + r + 8 * kh;
            const float e = __expf(acc[t][r] * TAU_INV);
            const size_t off = (size_t)i * N_ROWS + j;
            const float p = pos[off];
            const float q = neg[off];
            sp = fmaf(e, p, sp);
            st = fmaf(e, p + q, st);
        }
    }

    // deterministic reduction: shuffle within wave, LDS across waves
#pragma unroll
    for (int o = 16; o > 0; o >>= 1) {
        sp += __shfl_xor(sp, o, 32);
        st += __shfl_xor(st, o, 32);
    }
    if (lane == 0) { red[wave] = sp; red[8 + wave] = st; }
    __syncthreads();
    if (threadIdx.x == 0) {
        float a = 0.0f, b = 0.0f;
        for (int w = 0; w < 8; ++w) { a += red[w]; b += red[8 + w]; }
        const int blk = blockIdx.y * gridDim.x + blockIdx.x;
        partials[2 * blk]     = a;
        partials[2 * blk + 1] = b;
    }
}

// ---------------------------------------------------------------------------
// Kernel 4: final fixed-order reduction;  out = log(st) - log(sp)
// ---------------------------------------------------------------------------
__global__ __launch_bounds__(256) void finalize_kernel(
    const float* __restrict__ partials, int nblk, float* __restrict__ out)
{
    __shared__ float red[16];
    float sp = 0.0f, st = 0.0f;
    for (int i = threadIdx.x; i < nblk; i += 256) {
        sp += partials[2 * i];
        st += partials[2 * i + 1];
    }
#pragma unroll
    for (int o = 16; o > 0; o >>= 1) {
        sp += __shfl_xor(sp, o, 32);
        st += __shfl_xor(st, o, 32);
    }
    const int wave = threadIdx.x >> 5;
    const int lane = threadIdx.x & 31;
    if (lane == 0) { red[wave] = sp; red[8 + wave] = st; }
    __syncthreads();
    if (threadIdx.x == 0) {
        float a = 0.0f, b = 0.0f;
        for (int w = 0; w < 8; ++w) { a += red[w]; b += red[8 + w]; }
        out[0] = logf(b) - logf(a);
    }
}

// ---------------------------------------------------------------------------
extern "C" void kernel_launch(void* const* d_in, const int* in_sizes, int n_in,
                              void* d_out, int out_size, void* d_ws, size_t ws_size,
                              hipStream_t stream)
{
    (void)in_sizes; (void)n_in; (void)out_size; (void)ws_size;

    const float* v1  = (const float*)d_in[0];
    const float* v2  = (const float*)d_in[1];
    const float* pos = (const float*)d_in[2];
    const float* neg = (const float*)d_in[3];
    const float* W   = (const float*)d_in[4];
    const float* b   = (const float*)d_in[5];
    float* out = (float*)d_out;

    // workspace layout
    bf16_t* Wb   = (bf16_t*)d_ws;                       // 512*512
    bf16_t* Vb   = Wb + (size_t)DIM * DIM;              // 8192*512
    bf16_t* Zb   = Vb + (size_t)M_ALL * DIM;            // 8192*512
    float*  Zssq = (float*)(Zb + (size_t)M_ALL * DIM);  // 8192*4
    float*  partials = Zssq + (size_t)M_ALL * 4;        // 2048

    // 0) convert W, v1, v2 to bf16
    cvt_bf16_kernel<<<(DIM * DIM / 4 + 255) / 256, 256, 0, stream>>>(W, Wb, DIM * DIM / 4);
    cvt_bf16_kernel<<<(N_ROWS * DIM / 4 + 255) / 256, 256, 0, stream>>>(
        v1, Vb, N_ROWS * DIM / 4);
    cvt_bf16_kernel<<<(N_ROWS * DIM / 4 + 255) / 256, 256, 0, stream>>>(
        v2, Vb + (size_t)N_ROWS * DIM, N_ROWS * DIM / 4);

    // 1) z = ELU(v @ W^T + b), bf16 out + row sumsq partials
    gemm_bias_elu_kernel<<<dim3(DIM / 128, M_ALL / 128), 256, 0, stream>>>(
        Vb, Wb, b, Zb, Zssq);

    // 2) row-normalize in place
    norm_scale_kernel<<<M_ALL, 128, 0, stream>>>(Zb, Zssq);

    // 3) fused similarity + masked reduction
    bf16_t* z1 = Zb;
    bf16_t* z2 = Zb + (size_t)N_ROWS * DIM;
    sim_reduce_kernel<<<dim3(N_ROWS / 128, N_ROWS / 128), 256, 0, stream>>>(
        z1, z2, pos, neg, partials);

    // 4) scalar result
    finalize_kernel<<<1, 256, 0, stream>>>(partials, (N_ROWS / 128) * (N_ROWS / 128), out);
}
